// GATConvolutionLin_72911364817011
// MI455X (gfx1250) — matrix operand order repaired and checked by reading, not compiled, not run on hardware
//
#include <hip/hip_runtime.h>

// ---------------------------------------------------------------------------
// 2-layer GAT + linear head for MI455X (gfx1250, wave32, WMMA bf16).
// GEMMs run on v_wmma_f32_16x16x32_bf16 with fully register-resident
// accumulators (column-tile count is a template constant so the tile loop
// unrolls and nothing spills to scratch). Softmax/aggregation stay f32.
// ---------------------------------------------------------------------------

#define N_FEAT 128           // F_IN == HID == 128
#define N_CLS  40
#define N_CLS_PAD 48         // 3 x 16 WMMA column tiles

typedef __attribute__((ext_vector_type(16))) __bf16 v16bf;
typedef __attribute__((ext_vector_type(8)))  __bf16 v8bf;
typedef __attribute__((ext_vector_type(8)))  float  v8f;

// ---------------- helpers ----------------
__device__ __forceinline__ unsigned f2key(float f) {
  unsigned b = __float_as_uint(f);
  return (b & 0x80000000u) ? ~b : (b | 0x80000000u);   // order-preserving map
}
__device__ __forceinline__ float key2f(unsigned k) {
  unsigned b = (k & 0x80000000u) ? (k & 0x7FFFFFFFu) : ~k;
  return __uint_as_float(b);
}

// ---------------- conversion kernels ----------------
__global__ void conv_x_bf16(const float* __restrict__ x, __bf16* __restrict__ xb, long n) {
  long i = (long)blockIdx.x * blockDim.x + threadIdx.x;
  if (i < n) xb[i] = (__bf16)x[i];
}

// W row-major [128,128] (W[k][n]) -> bf16 column-major Wc[n*128+k]
__global__ void conv_w_colmajor(const float* __restrict__ W, __bf16* __restrict__ Wc) {
  int i = blockIdx.x * blockDim.x + threadIdx.x;          // i = n*128 + k
  if (i < N_FEAT * N_FEAT) {
    int n = i >> 7, k = i & 127;
    Wc[i] = (__bf16)W[k * N_FEAT + n];
  }
}

// Wl row-major [128,40] -> bf16 column-major padded [48][128]
__global__ void conv_wl_colmajor(const float* __restrict__ Wl, __bf16* __restrict__ Wlc) {
  int i = blockIdx.x * blockDim.x + threadIdx.x;          // i = n*128 + k
  if (i < N_CLS_PAD * N_FEAT) {
    int n = i >> 7, k = i & 127;
    Wlc[i] = (n < N_CLS) ? (__bf16)Wl[k * N_CLS + n] : (__bf16)0.0f;
  }
}

__global__ void zero_u32(unsigned* __restrict__ p, long n) {
  long i = (long)blockIdx.x * blockDim.x + threadIdx.x;
  if (i < n) p[i] = 0u;
}

// ---------------- WMMA GEMM: H[M,NT*16] = Xb[M,128] * Wc (col-major bf16) --
// One wave -> 16 rows x NT*16 columns.  K = 128 in four k0-steps of 32.
// NT is a compile-time constant so acc[] stays fully register-resident.
template <int NT>
__global__ __launch_bounds__(256) void gemm_bf16_wmma(
    const __bf16* __restrict__ Xb,   // [M,128] row-major bf16
    const __bf16* __restrict__ Wc,   // [NT*16,128] col-major bf16
    float* __restrict__ H,           // [M,NT*16] row-major f32
    int M) {
  int wave = (blockIdx.x * blockDim.x + threadIdx.x) >> 5;   // wave-uniform
  int lane = threadIdx.x & 31;
  int row0 = wave * 16;
  if (row0 >= M) return;                                     // uniform exit, EXEC stays full
  const int half = lane >> 4;      // 0 | 1
  const int l16  = lane & 15;
  const int NCOLS = NT * 16;

  v8f acc[NT];
#pragma unroll
  for (int t = 0; t < NT; ++t) acc[t] = (v8f){0.f,0.f,0.f,0.f,0.f,0.f,0.f,0.f};

  const __bf16* arowbase = Xb + (size_t)(row0 + l16) * N_FEAT;
  const __bf16* bbase    = Wc + (size_t)l16 * N_FEAT;

#pragma unroll
  for (int k0 = 0; k0 < N_FEAT; k0 += 32) {
    // ---- A fragment: row = row0+l16, K = k0 + 16*half + 0..15 (contig) ----
    const __bf16* arow = arowbase + k0 + half * 16;
    v8bf alo = *(const v8bf*)(arow);          // 16B load
    v8bf ahi = *(const v8bf*)(arow + 8);      // 16B load
    v16bf afrag = __builtin_shufflevector(alo, ahi,
        0,1,2,3,4,5,6,7,8,9,10,11,12,13,14,15);

#pragma unroll
    for (int t = 0; t < NT; ++t) {
      // ---- B fragment: col = t*16+l16, K = k0 + 16*half + 0..15 (contig) ----
      const __bf16* bcol = bbase + (size_t)t * 16 * N_FEAT + k0 + half * 16;
      v16bf bfrag = *(const v16bf*)bcol;      // 32B contiguous load (2 x b128)
      acc[t] = __builtin_amdgcn_wmma_f32_16x16x32_bf16(
          /*neg_a=*/false, afrag, /*neg_b=*/false, bfrag,
          /*c_mod=*/(short)0, acc[t], /*reuse_a=*/false, /*reuse_b=*/false);
    }
  }

  // ---- store D: lane holds N = t*16+l16, M = row0 + r + 8*half ----
#pragma unroll
  for (int t = 0; t < NT; ++t) {
#pragma unroll
    for (int r = 0; r < 8; ++r) {
      int row = row0 + r + 8 * half;
      H[(size_t)row * NCOLS + t * 16 + l16] = acc[t][r];
    }
  }
}

// ---------------- per-node score dots: s = h . a  (one wave per row) -------
__global__ void rowdots(const float* __restrict__ h,
                        const float* __restrict__ a_src,
                        const float* __restrict__ a_dst,
                        float* __restrict__ s_src, float* __restrict__ s_dst, int M) {
  int row  = (blockIdx.x * blockDim.x + threadIdx.x) >> 5;
  int lane = threadIdx.x & 31;
  if (row >= M) return;
  float4 v  = ((const float4*)(h + (size_t)row * N_FEAT))[lane];
  float4 as = ((const float4*)a_src)[lane];
  float4 ad = ((const float4*)a_dst)[lane];
  float ps = v.x*as.x + v.y*as.y + v.z*as.z + v.w*as.w;
  float pd = v.x*ad.x + v.y*ad.y + v.z*ad.z + v.w*ad.w;
#pragma unroll
  for (int m = 16; m >= 1; m >>= 1) {
    ps += __shfl_xor(ps, m, 32);
    pd += __shfl_xor(pd, m, 32);
  }
  if (lane == 0) { s_src[row] = ps; s_dst[row] = pd; }
}

// ---------------- edge pass 1: segment max (monotonic-key u32 atomic) ------
__global__ void edge_max(const int* __restrict__ ei, int E, int Eext,
                         const float* __restrict__ s_src, const float* __restrict__ s_dst,
                         unsigned* __restrict__ mkey) {
  int e = blockIdx.x * blockDim.x + threadIdx.x;
  if (e >= Eext) return;
  int src, dst;
  if (e < E) { src = ei[e]; dst = ei[E + e]; } else { src = dst = e - E; }
  float v = s_src[src] + s_dst[dst];
  v = (v < 0.f) ? 0.2f * v : v;                       // leaky_relu(0.2)
  atomicMax(&mkey[dst], f2key(v));
}

// ---------------- edge pass 2: exp + segment sum ---------------------------
__global__ void edge_exp(const int* __restrict__ ei, int E, int Eext,
                         const float* __restrict__ s_src, const float* __restrict__ s_dst,
                         const unsigned* __restrict__ mkey,
                         float* __restrict__ exbuf, float* __restrict__ denom) {
  int e = blockIdx.x * blockDim.x + threadIdx.x;
  if (e >= Eext) return;
  int src, dst;
  if (e < E) { src = ei[e]; dst = ei[E + e]; } else { src = dst = e - E; }
  float v = s_src[src] + s_dst[dst];
  v = (v < 0.f) ? 0.2f * v : v;
  float ex = __expf(v - key2f(mkey[dst]));
  exbuf[e] = ex;
  atomicAdd(&denom[dst], ex);
}

// ---------------- edge pass 3: weighted scatter (one wave per edge) --------
__global__ __launch_bounds__(256) void edge_scatter(
    const int* __restrict__ ei, int E, int Eext,
    const float* __restrict__ exbuf, const float* __restrict__ denom,
    const float* __restrict__ h, float* __restrict__ agg) {
  int e    = (blockIdx.x * blockDim.x + threadIdx.x) >> 5;
  int lane = threadIdx.x & 31;
  if (e >= Eext) return;
  int src, dst;
  if (e < E) { src = ei[e]; dst = ei[E + e]; } else { src = dst = e - E; }
  float alpha = exbuf[e] / denom[dst];
  float4 v = ((const float4*)(h + (size_t)src * N_FEAT))[lane];
  float* ad = agg + (size_t)dst * N_FEAT + lane * 4;
  atomicAdd(ad + 0, alpha * v.x);
  atomicAdd(ad + 1, alpha * v.y);
  atomicAdd(ad + 2, alpha * v.z);
  atomicAdd(ad + 3, alpha * v.w);
}

// ---------------- finalize: + bias, (relu), re-quantize to bf16 input ------
__global__ void finalize_layer(const float* __restrict__ agg, const float* __restrict__ b,
                               __bf16* __restrict__ xb, int do_relu, long n) {
  long i = (long)blockIdx.x * blockDim.x + threadIdx.x;
  if (i >= n) return;
  float y = agg[i] + b[i & (N_FEAT - 1)];
  if (do_relu) y = fmaxf(y, 0.f);
  xb[i] = (__bf16)y;
}

// ---------------- classifier log-softmax (one thread per row) --------------
__global__ void cls_logsoftmax(const float* __restrict__ raw /*[M,48]*/,
                               const float* __restrict__ bl,
                               float* __restrict__ out, int M) {
  int row = blockIdx.x * blockDim.x + threadIdx.x;
  if (row >= M) return;
  float v[N_CLS];
  float mx = -3.402823466e38f;
#pragma unroll
  for (int c = 0; c < N_CLS; ++c) {
    v[c] = raw[(size_t)row * N_CLS_PAD + c] + bl[c];
    mx = fmaxf(mx, v[c]);
  }
  float s = 0.f;
#pragma unroll
  for (int c = 0; c < N_CLS; ++c) s += __expf(v[c] - mx);
  float ls = __logf(s) + mx;
#pragma unroll
  for (int c = 0; c < N_CLS; ++c) out[(size_t)row * N_CLS + c] = v[c] - ls;
}

// ---------------- pass edge_index through to output tail -------------------
__global__ void copy_edges(const int* __restrict__ ei, int n, int* __restrict__ tail) {
  int i = blockIdx.x * blockDim.x + threadIdx.x;
  if (i < n) tail[i] = ei[i];
}

// ===========================================================================
extern "C" void kernel_launch(void* const* d_in, const int* in_sizes, int n_in,
                              void* d_out, int out_size, void* d_ws, size_t ws_size,
                              hipStream_t stream) {
  const float* x      = (const float*)d_in[0];
  const int*   ei     = (const int*)  d_in[1];
  const float* W1     = (const float*)d_in[2];
  const float* a1_src = (const float*)d_in[3];
  const float* a1_dst = (const float*)d_in[4];
  const float* b1     = (const float*)d_in[5];
  const float* W2     = (const float*)d_in[6];
  const float* a2_src = (const float*)d_in[7];
  const float* a2_dst = (const float*)d_in[8];
  const float* b2     = (const float*)d_in[9];
  const float* Wl     = (const float*)d_in[10];
  const float* bl     = (const float*)d_in[11];

  const int N    = in_sizes[0] / N_FEAT;   // 50000 nodes
  const int E    = in_sizes[1] / 2;        // 600000 edges
  const int Eext = E + N;                  // + self loops
  const long NF  = (long)N * N_FEAT;

  // ---- carve workspace ----
  char* ws = (char*)d_ws;
  size_t off = 0;
  auto carve = [&](size_t bytes) -> void* {
    void* p = ws + off;
    off += (bytes + 255) & ~(size_t)255;
    return p;
  };
  __bf16*   xb    = (__bf16*)  carve((size_t)NF * 2);
  float*    h     = (float*)   carve((size_t)NF * 4);
  float*    agg   = (float*)   carve((size_t)NF * 4);
  float*    s_src = (float*)   carve((size_t)N * 4);
  float*    s_dst = (float*)   carve((size_t)N * 4);
  unsigned* mkey  = (unsigned*)carve((size_t)N * 4);
  float*    denom = (float*)   carve((size_t)N * 4);
  float*    exbuf = (float*)   carve((size_t)Eext * 4);
  __bf16*   Wc1   = (__bf16*)  carve((size_t)N_FEAT * N_FEAT * 2);
  __bf16*   Wc2   = (__bf16*)  carve((size_t)N_FEAT * N_FEAT * 2);
  __bf16*   Wlc   = (__bf16*)  carve((size_t)N_CLS_PAD * N_FEAT * 2);
  float*    raw   = (float*)   carve((size_t)N * N_CLS_PAD * 4);
  (void)ws_size; (void)n_in; (void)out_size;

  const int B = 256;
  auto blocks = [](long n, int b) { return (int)((n + b - 1) / b); };

  // ---- one-time conversions ----
  conv_x_bf16<<<blocks(NF, B), B, 0, stream>>>(x, xb, NF);
  conv_w_colmajor<<<blocks(N_FEAT * N_FEAT, B), B, 0, stream>>>(W1, Wc1);
  conv_w_colmajor<<<blocks(N_FEAT * N_FEAT, B), B, 0, stream>>>(W2, Wc2);
  conv_wl_colmajor<<<blocks(N_CLS_PAD * N_FEAT, B), B, 0, stream>>>(Wl, Wlc);

  const int nwaves_gemm = (N + 15) / 16;                 // 3125 waves
  const int gemm_blocks = blocks((long)nwaves_gemm * 32, B);
  const int edge_blocks = blocks(Eext, B);
  const int scat_blocks = blocks((long)Eext * 32, B);

  for (int L = 0; L < 2; ++L) {
    const __bf16* Wc = (L == 0) ? Wc1 : Wc2;
    const float*  as = (L == 0) ? a1_src : a2_src;
    const float*  ad = (L == 0) ? a1_dst : a2_dst;
    const float*  bb = (L == 0) ? b1 : b2;

    gemm_bf16_wmma<8><<<gemm_blocks, B, 0, stream>>>(xb, Wc, h, N);
    rowdots<<<blocks((long)N * 32, B), B, 0, stream>>>(h, as, ad, s_src, s_dst, N);

    zero_u32<<<blocks(N, B), B, 0, stream>>>(mkey, N);
    zero_u32<<<blocks(N, B), B, 0, stream>>>((unsigned*)denom, N);
    zero_u32<<<blocks(NF, B), B, 0, stream>>>((unsigned*)agg, NF);

    edge_max<<<edge_blocks, B, 0, stream>>>(ei, E, Eext, s_src, s_dst, mkey);
    edge_exp<<<edge_blocks, B, 0, stream>>>(ei, E, Eext, s_src, s_dst, mkey, exbuf, denom);
    edge_scatter<<<scat_blocks, B, 0, stream>>>(ei, E, Eext, exbuf, denom, h, agg);

    finalize_layer<<<blocks(NF, B), B, 0, stream>>>(agg, bb, xb, /*relu=*/(L == 0), NF);
  }

  // ---- classifier GEMM (padded to 48 cols) + log_softmax ----
  gemm_bf16_wmma<3><<<gemm_blocks, B, 0, stream>>>(xb, Wlc, raw, N);
  float* outF = (float*)d_out;
  cls_logsoftmax<<<blocks(N, B), B, 0, stream>>>(raw, bl, outF, N);

  // ---- edge_index passthrough into output tail ----
  copy_edges<<<blocks(2 * E, B), B, 0, stream>>>(ei, 2 * E, (int*)(outF + (size_t)N * N_CLS));
}
